// TemplPairformerBlock_7164005450354
// MI455X (gfx1250) — matrix operand order repaired and checked by reading, not compiled, not run on hardware
//
#include <hip/hip_runtime.h>

typedef __attribute__((ext_vector_type(16))) __bf16 v16bf;
typedef __attribute__((ext_vector_type(8)))  __bf16 v8bf;
typedef __attribute__((ext_vector_type(8)))  float  v8f;
typedef unsigned int uint32x4 __attribute__((ext_vector_type(4)));
typedef int          int32x8  __attribute__((ext_vector_type(8)));
typedef int          int32x4  __attribute__((ext_vector_type(4)));

#define DEVINL static __device__ __forceinline__

DEVINL float sigmoidf_(float x) { return 1.0f / (1.0f + __expf(-x)); }

DEVINL v8f wmma_bf16(v16bf a, v16bf b, v8f c) {
  return __builtin_amdgcn_wmma_f32_16x16x32_bf16(false, a, false, b, (short)0, c,
                                                 false, false);
}

// A operand: 16x32 bf16 row-major tile (ld in elements).
// lanes 0-15: M=lane, K=0..15; lanes 16-31: M=lane-16, K=16..31
DEVINL v16bf load_a16x32(const __bf16* base, int ld) {
  int lane = threadIdx.x & 31;
  int r = lane & 15, hf = lane >> 4;
  const __bf16* p = base + r * ld + hf * 16;
  v8bf lo = *(const v8bf*)(p);
  v8bf hi = *(const v8bf*)(p + 8);
  v16bf out;
#pragma unroll
  for (int t = 0; t < 8; t++) { out[t] = lo[t]; out[t + 8] = hi[t]; }
  return out;
}

// A operand with only K=0..15 valid (upper 16 zero) — for DH=16 attention.
DEVINL v16bf load_a16x16pad(const __bf16* base, int ld) {
  int lane = threadIdx.x & 31;
  int r = lane & 15, hf = lane >> 4;
  const __bf16* p = base + r * ld;
  v8bf lo = *(const v8bf*)(p);
  v8bf hi = *(const v8bf*)(p + 8);
  v16bf out;
#pragma unroll
  for (int t = 0; t < 8; t++) { out[t] = hf ? (__bf16)0.0f : lo[t]; out[t + 8] = hf ? (__bf16)0.0f : hi[t]; }
  return out;
}

// B operand: 32x16 bf16, stored transposed Bt[n][k] (ld in elements).
// lanes 0-15: N=lane, K=0..15; lanes 16-31: N=lane-16, K=16..31
DEVINL v16bf load_bT16x32(const __bf16* baseT, int ld) {
  int lane = threadIdx.x & 31;
  int n = lane & 15, hf = lane >> 4;
  return *(const v16bf*)(baseT + n * ld + hf * 16);
}

// B operand with only K=0..15 valid (upper half-wave zeroed).
DEVINL v16bf load_bT16x16pad(const __bf16* baseT, int ld) {
  int lane = threadIdx.x & 31;
  int n = lane & 15, hf = lane >> 4;
  v16bf t = *(const v16bf*)(baseT + n * ld);
  if (hf) {
#pragma unroll
    for (int e = 0; e < 16; e++) t[e] = (__bf16)0.0f;
  }
  return t;
}

// 16 positions x 64 outputs, K=64 (weights Wt[n][k], ld=64)
DEVINL void gemm16x64(const v16bf& A0, const v16bf& A1, const __bf16* WT,
                      v8f out[4]) {
#pragma unroll
  for (int nt = 0; nt < 4; nt++) {
    v8f acc = {};
    v16bf B0 = load_bT16x32(WT + nt * 16 * 64, 64);
    acc = wmma_bf16(A0, B0, acc);
    v16bf B1 = load_bT16x32(WT + nt * 16 * 64 + 32, 64);
    acc = wmma_bf16(A1, B1, acc);
    out[nt] = acc;
  }
}

// ---- Tensor Data Mover: 2-D tile (tile_d0 x tile_d1 elems, 2B) -> LDS -------
// D# per CDNA5 ISA 8.3/8.4: group0={count,lds,global,type=2}, group1 packs
// data_size=2B, tensor dims (large: no OOB), tile dims, dim0 stride.
DEVINL void tdm_load_2d(unsigned lds_addr, const void* gaddr, unsigned tile_d0,
                        unsigned tile_d1, unsigned stride_elems) {
  unsigned long long ga = (unsigned long long)gaddr;
  uint32x4 g0;
  g0[0] = 1u;                                   // count=1 (valid), user mode
  g0[1] = lds_addr;                             // LDS byte address
  g0[2] = (unsigned)ga;                         // global_addr[31:0]
  g0[3] = (unsigned)(ga >> 32) | (2u << 30);    // global_addr[56:32] | type=2
  const unsigned td0 = 1u << 20, td1 = 1u << 20;  // tensor dims (no OOB)
  int32x8 g1;
  g1[0] = (int)(1u << 16);                      // data_size=2B, wg_mask=0
  g1[1] = (int)((td0 & 0xFFFFu) << 16);         // tensor_dim0[15:0]
  g1[2] = (int)((td0 >> 16) | ((td1 & 0xFFFFu) << 16));
  g1[3] = (int)((td1 >> 16) | (tile_d0 << 16)); // tile_dim0
  g1[4] = (int)tile_d1;                         // tile_dim1 (tile_dim2=0)
  g1[5] = (int)stride_elems;                    // tensor_dim0_stride[31:0]
  g1[6] = 0;
  g1[7] = 0;
  int32x4 z4 = {0, 0, 0, 0};                    // 2-D tensor: groups 2/3 unused
  int32x8 z8 = {0, 0, 0, 0, 0, 0, 0, 0};
  __builtin_amdgcn_tensor_load_to_lds(g0, g1, z4, z4, z8, 0);
}

DEVINL unsigned lds_off(const void* p) {  // low 32 bits of flat addr = LDS addr
  return (unsigned)(unsigned long long)p;
}

// ---------------- weight conversion: f32 W[K][N] -> bf16 Wt[n][k] -------------
__global__ void wconv_kernel(const float* __restrict__ src,
                             __bf16* __restrict__ dst, int K, int N) {
  int idx = blockIdx.x * blockDim.x + threadIdx.x;
  if (idx >= K * N) return;
  int k = idx / N, n = idx - k * N;
  dst[n * K + k] = (__bf16)src[idx];
}

// ---------------- LayerNorm over 64 channels -> bf16 ------------------------
__global__ __launch_bounds__(256) void ln_kernel(const float* __restrict__ Z,
                                                 const float* __restrict__ g,
                                                 const float* __restrict__ b,
                                                 __bf16* __restrict__ zn,
                                                 int transpose) {
  int p = blockIdx.x * 256 + threadIdx.x;  // 65536 positions
  const float* zp = Z + (size_t)p * 64;
  float x[64];
#pragma unroll
  for (int c = 0; c < 64; c++) x[c] = zp[c];
  float m = 0.0f;
#pragma unroll
  for (int c = 0; c < 64; c++) m += x[c];
  m *= (1.0f / 64.0f);
  float v = 0.0f;
#pragma unroll
  for (int c = 0; c < 64; c++) { float d = x[c] - m; v += d * d; }
  v *= (1.0f / 64.0f);
  float r = rsqrtf(v + 1e-5f);
  int i = p >> 8, j = p & 255;
  int q = transpose ? (j * 256 + i) : p;
  __bf16* o = zn + (size_t)q * 64;
#pragma unroll
  for (int c = 0; c < 64; c++) o[c] = (__bf16)((x[c] - m) * r * g[c] + b[c]);
}

// ---------------- tri-mul: gated a/b/g projections --------------------------
__global__ __launch_bounds__(256) void trimul_proj(
    const __bf16* __restrict__ zn, const __bf16* __restrict__ wp,
    const float* __restrict__ ag_b, const float* __restrict__ ap_b,
    const float* __restrict__ bg_b, const float* __restrict__ bp_b,
    const float* __restrict__ g_b, __bf16* __restrict__ aT,
    __bf16* __restrict__ bT, float* __restrict__ gbuf, int transposeAB) {
  int w = threadIdx.x >> 5, lane = threadIdx.x & 31;
  int pt = blockIdx.x * 8 + w;
  int p0 = pt * 16;
  int nl = lane & 15, hf = lane >> 4;
  v16bf A0 = load_a16x32(zn + p0 * 64, 64);
  v16bf A1 = load_a16x32(zn + p0 * 64 + 32, 64);
  v8f s[4], t[4];

  // a = sigmoid(zn@ag + agb) * (zn@ap + apb)   -> aT[c][.,.]
  gemm16x64(A0, A1, wp + 0, s);
  gemm16x64(A0, A1, wp + 4096, t);
#pragma unroll
  for (int nt = 0; nt < 4; nt++)
#pragma unroll
    for (int e = 0; e < 8; e++) {
      int n = nt * 16 + nl, m = e + hf * 8, pos = p0 + m;
      int i = pos >> 8, j = pos & 255;
      int di = transposeAB ? (j * 256 + i) : pos;
      float va = sigmoidf_(s[nt][e] + ag_b[n]) * (t[nt][e] + ap_b[n]);
      aT[n * 65536 + di] = (__bf16)va;
    }
  // b
  gemm16x64(A0, A1, wp + 8192, s);
  gemm16x64(A0, A1, wp + 12288, t);
#pragma unroll
  for (int nt = 0; nt < 4; nt++)
#pragma unroll
    for (int e = 0; e < 8; e++) {
      int n = nt * 16 + nl, m = e + hf * 8, pos = p0 + m;
      int i = pos >> 8, j = pos & 255;
      int di = transposeAB ? (j * 256 + i) : pos;
      float vb = sigmoidf_(s[nt][e] + bg_b[n]) * (t[nt][e] + bp_b[n]);
      bT[n * 65536 + di] = (__bf16)vb;
    }
  // g
  gemm16x64(A0, A1, wp + 16384, s);
#pragma unroll
  for (int nt = 0; nt < 4; nt++)
#pragma unroll
    for (int e = 0; e < 8; e++) {
      int n = nt * 16 + nl, m = e + hf * 8, pos = p0 + m;
      gbuf[pos * 64 + n] = sigmoidf_(s[nt][e] + g_b[n]);
    }
}

// ---------------- tri-mul: per-channel 256x256x256 GEMM, TDM-staged ----------
// Block = one channel c, one 128x128 output tile. 8 waves in a 2x4 grid, each
// owning 64x32. K staged 128x32 per operand via tensor_load_to_lds, double-
// buffered (TENSORcnt), 8 WMMA per wave per K-step out of LDS.
__global__ __launch_bounds__(256) void trimul_gemm(
    const __bf16* __restrict__ aT, const __bf16* __restrict__ bT,
    float* __restrict__ xbuf) {
  __shared__ __bf16 As[2][128][32];  // 16 KB
  __shared__ __bf16 Bs[2][128][32];  // 16 KB
  int w = threadIdx.x >> 5, lane = threadIdx.x & 31;
  int nl = lane & 15, hf = lane >> 4;
  int c = blockIdx.x >> 2;
  int tile = blockIdx.x & 3;
  int i0 = (tile >> 1) * 128, j0 = (tile & 1) * 128;
  const __bf16* A = aT + c * 65536 + i0 * 256;
  const __bf16* B = bT + c * 65536 + j0 * 256;
  int wm = (w >> 2) * 64, wn = (w & 3) * 32;

  v8f acc[4][2] = {};
  if (w == 0) {
    tdm_load_2d(lds_off(&As[0][0][0]), A, 32, 128, 256);
    tdm_load_2d(lds_off(&Bs[0][0][0]), B, 32, 128, 256);
    __builtin_amdgcn_s_wait_tensorcnt(0);
  }
  __syncthreads();
#pragma unroll 1
  for (int s = 0; s < 8; s++) {
    int buf = s & 1;
    if (w == 0 && s < 7) {  // prefetch next K panel into other buffer
      tdm_load_2d(lds_off(&As[buf ^ 1][0][0]), A + (s + 1) * 32, 32, 128, 256);
      tdm_load_2d(lds_off(&Bs[buf ^ 1][0][0]), B + (s + 1) * 32, 32, 128, 256);
    }
    v16bf vb0 = load_bT16x32(&Bs[buf][wn][0], 32);
    v16bf vb1 = load_bT16x32(&Bs[buf][wn + 16][0], 32);
#pragma unroll
    for (int mt = 0; mt < 4; mt++) {
      v16bf va = load_a16x32(&As[buf][wm + mt * 16][0], 32);
      acc[mt][0] = wmma_bf16(va, vb0, acc[mt][0]);
      acc[mt][1] = wmma_bf16(va, vb1, acc[mt][1]);
    }
    if (w == 0 && s < 7) __builtin_amdgcn_s_wait_tensorcnt(0);
    __syncthreads();
  }
#pragma unroll
  for (int mt = 0; mt < 4; mt++)
#pragma unroll
    for (int nt = 0; nt < 2; nt++)
#pragma unroll
      for (int e = 0; e < 8; e++) {
        int ii = i0 + wm + mt * 16 + e + hf * 8;
        int jj = j0 + wn + nt * 16 + nl;
        xbuf[(ii * 256 + jj) * 64 + c] = acc[mt][nt][e];
      }
}

// ---------------- tri-mul: LN + gate + output projection + residual ----------
__global__ __launch_bounds__(256) void trimul_out(
    const float* __restrict__ xbuf, const float* __restrict__ gbuf,
    const float* __restrict__ lng, const float* __restrict__ lnb,
    const float* __restrict__ o_w, const float* __restrict__ o_b,
    float* __restrict__ Z) {
  __shared__ float xs[4][65];
  int t = threadIdx.x;
  int lp = t >> 6, c = t & 63;
  int pos = blockIdx.x * 4 + lp;
  xs[lp][c] = xbuf[(size_t)pos * 64 + c];
  __syncthreads();
  float m = 0.0f;
#pragma unroll
  for (int k = 0; k < 64; k++) m += xs[lp][k];
  m *= (1.0f / 64.0f);
  float v = 0.0f;
#pragma unroll
  for (int k = 0; k < 64; k++) { float d = xs[lp][k] - m; v += d * d; }
  v *= (1.0f / 64.0f);
  float r = rsqrtf(v + 1e-5f);
  float acc = 0.0f;
#pragma unroll
  for (int k = 0; k < 64; k++) {
    float xn = (xs[lp][k] - m) * r * lng[k] + lnb[k];
    acc += xn * o_w[k * 64 + c];
  }
  Z[(size_t)pos * 64 + c] += gbuf[(size_t)pos * 64 + c] * (acc + o_b[c]);
}

// ---------------- attention: q/k/v/g projections ----------------------------
__global__ __launch_bounds__(256) void attn_qkvg(
    const __bf16* __restrict__ zn, const __bf16* __restrict__ wp,
    __bf16* __restrict__ qbuf, __bf16* __restrict__ kbuf,
    __bf16* __restrict__ vT, float* __restrict__ gbuf) {
  int w = threadIdx.x >> 5, lane = threadIdx.x & 31;
  int pt = blockIdx.x * 8 + w;
  int p0 = pt * 16;
  int nl = lane & 15, hf = lane >> 4;
  v16bf A0 = load_a16x32(zn + p0 * 64, 64);
  v16bf A1 = load_a16x32(zn + p0 * 64 + 32, 64);
  v8f s[4];

  gemm16x64(A0, A1, wp + 0, s);  // q -> [h][i][j][d]
#pragma unroll
  for (int nt = 0; nt < 4; nt++)
#pragma unroll
    for (int e = 0; e < 8; e++) {
      int n = nt * 16 + nl, m = e + hf * 8, pos = p0 + m;
      int i = pos >> 8, j = pos & 255, head = n >> 4, d = n & 15;
      qbuf[((head * 256 + i) * 256 + j) * 16 + d] = (__bf16)s[nt][e];
    }
  gemm16x64(A0, A1, wp + 4096, s);  // k -> [h][i][k][d]
#pragma unroll
  for (int nt = 0; nt < 4; nt++)
#pragma unroll
    for (int e = 0; e < 8; e++) {
      int n = nt * 16 + nl, m = e + hf * 8, pos = p0 + m;
      int i = pos >> 8, j = pos & 255, head = n >> 4, d = n & 15;
      kbuf[((head * 256 + i) * 256 + j) * 16 + d] = (__bf16)s[nt][e];
    }
  gemm16x64(A0, A1, wp + 8192, s);  // v -> [h][i][d][k]  (B-operand layout)
#pragma unroll
  for (int nt = 0; nt < 4; nt++)
#pragma unroll
    for (int e = 0; e < 8; e++) {
      int n = nt * 16 + nl, m = e + hf * 8, pos = p0 + m;
      int i = pos >> 8, j = pos & 255, head = n >> 4, d = n & 15;
      vT[((head * 256 + i) * 16 + d) * 256 + j] = (__bf16)s[nt][e];
    }
  gemm16x64(A0, A1, wp + 12288, s);  // g (sigmoid) -> [i][j][h*16+d] f32
#pragma unroll
  for (int nt = 0; nt < 4; nt++)
#pragma unroll
    for (int e = 0; e < 8; e++) {
      int n = nt * 16 + nl, m = e + hf * 8, pos = p0 + m;
      gbuf[pos * 64 + n] = sigmoidf_(s[nt][e]);
    }
}

// ---------------- attention: pair bias b[h][j][k] ----------------------------
__global__ __launch_bounds__(256) void attn_bias_k(
    const __bf16* __restrict__ zn, const float* __restrict__ bw,
    float* __restrict__ bias) {
  int p = blockIdx.x * 256 + threadIdx.x;
  const __bf16* zp = zn + (size_t)p * 64;
  float a0 = 0, a1 = 0, a2 = 0, a3 = 0;
#pragma unroll
  for (int c = 0; c < 64; c++) {
    float z = (float)zp[c];
    a0 += z * bw[c * 4 + 0];
    a1 += z * bw[c * 4 + 1];
    a2 += z * bw[c * 4 + 2];
    a3 += z * bw[c * 4 + 3];
  }
  bias[0 * 65536 + p] = a0;
  bias[1 * 65536 + p] = a1;
  bias[2 * 65536 + p] = a2;
  bias[3 * 65536 + p] = a3;
}

// ---------------- attention core: scores -> softmax -> @V (fused, per wave) --
__global__ __launch_bounds__(128) void attn_core(
    const __bf16* __restrict__ qbuf, const __bf16* __restrict__ kbuf,
    const __bf16* __restrict__ vT, const float* __restrict__ bias,
    const float* __restrict__ gbuf, __bf16* __restrict__ obuf) {
  __shared__ float sc[4][16][256];  // 64 KB: per-wave 16 j-rows x 256 k scores
  int w = threadIdx.x >> 5, lane = threadIdx.x & 31;
  int gt = blockIdx.x * 4 + w;  // 16384 = h(4) x i(256) x jt(16)
  int jt = gt & 15;
  int i = (gt >> 4) & 255;
  int head = gt >> 12;
  int j0 = jt * 16;
  int nl = lane & 15, hf = lane >> 4;

  const __bf16* Q = qbuf + ((head * 256 + i) * 256 + j0) * 16;
  v16bf A = load_a16x16pad(Q, 16);
  const float* brow = bias + head * 65536;
#pragma unroll 1
  for (int kt = 0; kt < 16; kt++) {
    const __bf16* K = kbuf + ((head * 256 + i) * 256 + kt * 16) * 16;
    v16bf B = load_bT16x16pad(K, 16);
    v8f acc = {};
    acc = wmma_bf16(A, B, acc);
#pragma unroll
    for (int e = 0; e < 8; e++) {
      int m = e + hf * 8;
      int kk = kt * 16 + nl;
      sc[w][m][kk] = acc[e] * 0.25f + brow[(j0 + m) * 256 + kk];
    }
  }
  __builtin_amdgcn_wave_barrier();
  if (lane < 16) {  // one row per lane: softmax over k, write bf16 in place
    float* r = &sc[w][lane][0];
    float mx = -3.0e38f;
    for (int k = 0; k < 256; k++) mx = fmaxf(mx, r[k]);
    float sum = 0.0f;
    for (int k = 0; k < 256; k++) { float e = __expf(r[k] - mx); r[k] = e; sum += e; }
    float inv = 1.0f / sum;
    __bf16* pb = (__bf16*)r;
    for (int k = 0; k < 256; k++) { float e = r[k]; pb[k] = (__bf16)(e * inv); }
  }
  __builtin_amdgcn_wave_barrier();
  const __bf16* P = (const __bf16*)&sc[w][0][0];  // row pitch 512 bf16
  const __bf16* V = vT + (head * 256 + i) * 16 * 256;
  v8f oacc = {};
#pragma unroll 1
  for (int k = 0; k < 256; k += 32) {
    v16bf a = load_a16x32(P + k, 512);  // LDS A-operand
    v16bf b = load_bT16x32(V + k, 256);
    oacc = wmma_bf16(a, b, oacc);
  }
#pragma unroll
  for (int e = 0; e < 8; e++) {
    int m = e + hf * 8;
    int j = j0 + m;
    int c = head * 16 + nl;
    float gv = gbuf[(i * 256 + j) * 64 + c];
    obuf[(i * 256 + j) * 64 + c] = (__bf16)(gv * oacc[e]);
  }
}

// ---------------- attention: output projection + residual --------------------
__global__ __launch_bounds__(256) void attn_out(
    const __bf16* __restrict__ obuf, const __bf16* __restrict__ oWT,
    const float* __restrict__ o_b, float* __restrict__ Z, int transpose) {
  int w = threadIdx.x >> 5, lane = threadIdx.x & 31;
  int pt = blockIdx.x * 8 + w;
  int p0 = pt * 16;
  int nl = lane & 15, hf = lane >> 4;
  v16bf A0 = load_a16x32(obuf + p0 * 64, 64);
  v16bf A1 = load_a16x32(obuf + p0 * 64 + 32, 64);
  v8f s[4];
  gemm16x64(A0, A1, oWT, s);
#pragma unroll
  for (int nt = 0; nt < 4; nt++)
#pragma unroll
    for (int e = 0; e < 8; e++) {
      int n = nt * 16 + nl, m = e + hf * 8, pos = p0 + m;
      int i = pos >> 8, j = pos & 255;
      int dp = transpose ? (j * 256 + i) : pos;
      Z[dp * 64 + n] += s[nt][e] + o_b[n];
    }
}

// ---------------- transition: silu(zn@w1)*(zn@w2) @ w3 + residual ------------
__global__ __launch_bounds__(128) void trans_kernel(
    const __bf16* __restrict__ zn, const __bf16* __restrict__ w1T,
    const __bf16* __restrict__ w2T, const __bf16* __restrict__ w3T,
    float* __restrict__ Z) {
  __shared__ __bf16 hb[4][16][256];  // 32 KB
  int w = threadIdx.x >> 5, lane = threadIdx.x & 31;
  int pt = blockIdx.x * 4 + w;
  int p0 = pt * 16;
  int nl = lane & 15, hf = lane >> 4;
  v16bf A0 = load_a16x32(zn + p0 * 64, 64);
  v16bf A1 = load_a16x32(zn + p0 * 64 + 32, 64);
#pragma unroll 1
  for (int nt = 0; nt < 16; nt++) {
    v8f a1 = {}, a2 = {};
    v16bf B0 = load_bT16x32(w1T + nt * 16 * 64, 64);
    a1 = wmma_bf16(A0, B0, a1);
    v16bf B1 = load_bT16x32(w1T + nt * 16 * 64 + 32, 64);
    a1 = wmma_bf16(A1, B1, a1);
    B0 = load_bT16x32(w2T + nt * 16 * 64, 64);
    a2 = wmma_bf16(A0, B0, a2);
    B1 = load_bT16x32(w2T + nt * 16 * 64 + 32, 64);
    a2 = wmma_bf16(A1, B1, a2);
#pragma unroll
    for (int e = 0; e < 8; e++) {
      int m = e + hf * 8;
      float x = a1[e];
      hb[w][m][nt * 16 + nl] = (__bf16)((x * sigmoidf_(x)) * a2[e]);
    }
  }
  __builtin_amdgcn_wave_barrier();
  v8f acc[4] = {};
  const __bf16* H = &hb[w][0][0];
#pragma unroll 1
  for (int k = 0; k < 256; k += 32) {
    v16bf a = load_a16x32(H + k, 256);  // LDS A-operand
#pragma unroll
    for (int nt = 0; nt < 4; nt++) {
      v16bf b = load_bT16x32(w3T + nt * 16 * 256 + k, 256);
      acc[nt] = wmma_bf16(a, b, acc[nt]);
    }
  }
#pragma unroll
  for (int nt = 0; nt < 4; nt++)
#pragma unroll
    for (int e = 0; e < 8; e++) {
      int m = e + hf * 8, pos = p0 + m, n = nt * 16 + nl;
      Z[pos * 64 + n] += acc[nt][e];
    }
}

// ======================== host launcher =====================================
extern "C" void kernel_launch(void* const* d_in, const int* in_sizes, int n_in,
                              void* d_out, int out_size, void* d_ws,
                              size_t ws_size, hipStream_t stream) {
  (void)in_sizes; (void)n_in; (void)ws_size;
  const float* Zin = (const float*)d_in[0];
  float* Z = (float*)d_out;
  char* ws = (char*)d_ws;
  __bf16* wpool = (__bf16*)(ws + 0x0);        // 256 KB bf16 weights (transposed)
  __bf16* znb   = (__bf16*)(ws + 0x40000);    // 8 MB   LN output bf16
  __bf16* aT    = (__bf16*)(ws + 0x840000);   // 8 MB   a / q
  __bf16* bT    = (__bf16*)(ws + 0x1040000);  // 8 MB   b / vT
  __bf16* kb    = (__bf16*)(ws + 0x1840000);  // 8 MB   k
  __bf16* ob    = (__bf16*)(ws + 0x2040000);  // 8 MB   gated attn output
  float*  gbuf  = (float*)(ws + 0x2840000);   // 16 MB  gates f32
  float*  xbuf  = (float*)(ws + 0x3840000);   // 16 MB  tri-mul product f32
  float*  bias  = (float*)(ws + 0x4840000);   // 1 MB   attn pair bias

  auto W = [&](int idx) { return (const float*)d_in[idx]; };

  // Z working copy in d_out (inputs must not be mutated)
  (void)hipMemcpyAsync(Z, Zin, (size_t)out_size * sizeof(float),
                       hipMemcpyDeviceToDevice, stream);

  // bf16 transposed weight pool (element offsets)
  struct WC { int src; int off; int K; int N; };
  const WC wcs[] = {
      {3, 0, 64, 64},      {5, 4096, 64, 64},   {7, 8192, 64, 64},
      {9, 12288, 64, 64},  {13, 16384, 64, 64},                       // tmo
      {19, 20480, 64, 64}, {21, 24576, 64, 64}, {23, 28672, 64, 64},
      {25, 32768, 64, 64}, {29, 36864, 64, 64},                       // tmi
      {35, 40960, 64, 64}, {36, 45056, 64, 64}, {37, 49152, 64, 64},
      {39, 53248, 64, 64}, {40, 57344, 64, 64},                       // tas
      {44, 61440, 64, 64}, {45, 65536, 64, 64}, {46, 69632, 64, 64},
      {48, 73728, 64, 64}, {49, 77824, 64, 64},                       // tae
      {53, 81920, 64, 256}, {54, 98304, 64, 256}, {55, 114688, 256, 64},  // trans
  };
  for (const auto& wc : wcs) {
    int total = wc.K * wc.N;
    wconv_kernel<<<(total + 255) / 256, 256, 0, stream>>>(
        W(wc.src), wpool + wc.off, wc.K, wc.N);
  }

  // ---- tri-mul outgoing (tmo: idx 1..16) ----
  ln_kernel<<<256, 256, 0, stream>>>(Z, W(1), W(2), znb, 0);
  trimul_proj<<<512, 256, 0, stream>>>(znb, wpool + 0, W(4), W(6), W(8), W(10),
                                       W(14), aT, bT, gbuf, 0);
  trimul_gemm<<<256, 256, 0, stream>>>(aT, bT, xbuf);
  trimul_out<<<16384, 256, 0, stream>>>(xbuf, gbuf, W(11), W(12), W(15), W(16), Z);

  // ---- tri-mul incoming (tmi: idx 17..32) ----
  ln_kernel<<<256, 256, 0, stream>>>(Z, W(17), W(18), znb, 0);
  trimul_proj<<<512, 256, 0, stream>>>(znb, wpool + 20480, W(20), W(22), W(24),
                                       W(26), W(30), aT, bT, gbuf, 1);
  trimul_gemm<<<256, 256, 0, stream>>>(aT, bT, xbuf);
  trimul_out<<<16384, 256, 0, stream>>>(xbuf, gbuf, W(27), W(28), W(31), W(32), Z);

  // ---- tri-attn starting node (tas: idx 33..41) ----
  ln_kernel<<<256, 256, 0, stream>>>(Z, W(33), W(34), znb, 0);
  attn_qkvg<<<512, 256, 0, stream>>>(znb, wpool + 40960, aT, kb, bT, gbuf);
  attn_bias_k<<<256, 256, 0, stream>>>(znb, W(38), bias);
  attn_core<<<4096, 128, 0, stream>>>(aT, kb, bT, bias, gbuf, ob);
  attn_out<<<512, 256, 0, stream>>>(ob, wpool + 57344, W(41), Z, 0);

  // ---- tri-attn ending node (tae: idx 42..50), transposed frame ----
  ln_kernel<<<256, 256, 0, stream>>>(Z, W(42), W(43), znb, 1);
  attn_qkvg<<<512, 256, 0, stream>>>(znb, wpool + 61440, aT, kb, bT, gbuf);
  attn_bias_k<<<256, 256, 0, stream>>>(znb, W(47), bias);
  attn_core<<<4096, 128, 0, stream>>>(aT, kb, bT, bias, gbuf, ob);
  attn_out<<<512, 256, 0, stream>>>(ob, wpool + 77824, W(50), Z, 1);

  // ---- transition (idx 51..55) ----
  ln_kernel<<<256, 256, 0, stream>>>(Z, W(51), W(52), znb, 0);
  trans_kernel<<<1024, 128, 0, stream>>>(znb, wpool + 81920, wpool + 98304,
                                         wpool + 114688, Z);
}
